// _PointnetSAModuleBase_8323646620278
// MI455X (gfx1250) — matrix operand order repaired and checked by reading, not compiled, not run on hardware
//
#include <hip/hip_runtime.h>
#include <hip/hip_bf16.h>

// ---------------------------------------------------------------------------
// PointNet++ SA module for MI455X (gfx1250, wave32).
//  Kernel 1: ball query (one centroid per wave, ballot-based ordered pick),
//            also emits new_xyz and samp_idx.
//  Kernel 2: gather + 3-layer MLP via v_wmma_f32_16x16x32_f16 + max-pool.
// ---------------------------------------------------------------------------

typedef __attribute__((ext_vector_type(16))) _Float16 v16h;
typedef __attribute__((ext_vector_type(8)))  float    v8f;

#define BATCH    2
#define NPTS     16384
#define NCH      64
#define NPOINT   4096
#define NSAMP    32
#define RADIUS2  0.16f   // 0.4^2

// ---------------------------------------------------------------------------
// Kernel 1: ball query. One wave (32 lanes) per centroid.
// ---------------------------------------------------------------------------
__global__ __launch_bounds__(128) void ball_query_kernel(
    const float* __restrict__ xyz,      // [B, N, 3]
    float*       __restrict__ out_newxyz, // [B, S, 3]
    int*         __restrict__ out_sampidx,// [B, S]
    int*         __restrict__ idx_ws)     // [B*S, 32]
{
    const int lane = threadIdx.x & 31;
    const int wave = threadIdx.x >> 5;
    const int gid  = blockIdx.x * (blockDim.x >> 5) + wave;
    if (gid >= BATCH * NPOINT) return;
    const int b = gid / NPOINT;
    const int s = gid % NPOINT;

    const float cx = xyz[((size_t)b * NPTS + s) * 3 + 0];
    const float cy = xyz[((size_t)b * NPTS + s) * 3 + 1];
    const float cz = xyz[((size_t)b * NPTS + s) * 3 + 2];

    // new_xyz = xyz[:, :npoint]; samp_idx = arange
    if (lane < 3) out_newxyz[((size_t)b * NPOINT + s) * 3 + lane] =
                      xyz[((size_t)b * NPTS + s) * 3 + lane];
    if (lane == 0) out_sampidx[gid] = s;

    int  count = 0;
    int  first = 0;
    bool have_first = false;

    for (int base = 0; base < NPTS && count < NSAMP; base += 32) {
        const int p = base + lane;
        const float dx = xyz[((size_t)b * NPTS + p) * 3 + 0] - cx;
        const float dy = xyz[((size_t)b * NPTS + p) * 3 + 1] - cy;
        const float dz = xyz[((size_t)b * NPTS + p) * 3 + 2] - cz;
        const bool pred = (dx * dx + dy * dy + dz * dz) < RADIUS2;
        const unsigned mask = (unsigned)__ballot(pred);   // wave32: low 32 bits
        if (!have_first && mask) {
            first = base + (int)__builtin_ctz(mask);
            have_first = true;
        }
        if (pred) {
            const int rank = count + __popc(mask & ((1u << lane) - 1u));
            if (rank < NSAMP) idx_ws[(size_t)gid * NSAMP + rank] = p;
        }
        count += __popc(mask);
    }
    if (count > NSAMP) count = NSAMP;
    // pad empty slots with first hit (0 if no hit at all)
    if (lane >= count) idx_ws[(size_t)gid * NSAMP + lane] = have_first ? first : 0;
}

// ---------------------------------------------------------------------------
// WMMA fragment loaders (wave32; contiguous k order k = 16*half + e, same in A and B)
// ---------------------------------------------------------------------------
__device__ __forceinline__ v16h load_a_frag(const _Float16* buf, int stride,
                                            int m_tile, int k_tile, int lane) {
    const int m    = m_tile * 16 + (lane & 15);
    const int half = lane >> 4;
    const _Float16* p = buf + m * stride + k_tile * 32 + half * 16;
    v16h a;
#pragma unroll
    for (int e = 0; e < 16; ++e) a[e] = p[e];
    return a;
}

__device__ __forceinline__ v16h load_b_frag(const _Float16* wt, int kdim,
                                            int n_tile, int k_tile, int lane) {
    const int n    = n_tile * 16 + (lane & 15);
    const int half = lane >> 4;
    const _Float16* p = wt + n * kdim + k_tile * 32 + half * 16;
    v16h w;
#pragma unroll
    for (int e = 0; e < 16; ++e) w[e] = p[e];
    return w;
}

// ---------------------------------------------------------------------------
// Kernel 2: gather + MLP(67->64->64->128) + max over 32 neighbors.
// Block = 64 threads = 2 waves, one sample per wave.
// ---------------------------------------------------------------------------
__global__ __launch_bounds__(64) void group_mlp_kernel(
    const float* __restrict__ xyz,       // [B, N, 3]
    const float* __restrict__ features,  // [B, C, N]
    const float* __restrict__ W1, const float* __restrict__ b1, // [67,64],[64]
    const float* __restrict__ W2, const float* __restrict__ b2, // [64,64],[64]
    const float* __restrict__ W3, const float* __restrict__ b3, // [64,128],[128]
    const int*   __restrict__ idx_ws,    // [B*S, 32]
    float*       __restrict__ out_feat)  // [B, 128, S]
{
    __shared__ _Float16 Wt1[64 * 96];    // transposed, K padded 67->96
    __shared__ _Float16 Wt2[64 * 64];    // transposed
    __shared__ _Float16 Wt3[128 * 64];   // transposed
    __shared__ float    bias1[64], bias2[64], bias3[128];
    __shared__ _Float16 gbuf[2][32 * 96];  // layer-1 input, reused as h2
    __shared__ _Float16 hbuf[2][32 * 64];  // h1

    const int tid = threadIdx.x;

    // --- stage weights (transposed, f16) + biases into LDS ---
    for (int i = tid; i < 64 * 96; i += 64) {
        const int d = i / 96, c = i % 96;
        Wt1[i] = (c < 67) ? (_Float16)W1[c * 64 + d] : (_Float16)0.0f;
    }
    for (int i = tid; i < 64 * 64; i += 64) {
        const int d = i / 64, c = i % 64;
        Wt2[i] = (_Float16)W2[c * 64 + d];
    }
    for (int i = tid; i < 128 * 64; i += 64) {
        const int d = i / 64, c = i % 64;
        Wt3[i] = (_Float16)W3[c * 128 + d];
    }
    for (int i = tid; i < 64; i += 64) { bias1[i] = b1[i]; bias2[i] = b2[i]; }
    for (int i = tid; i < 128; i += 64) bias3[i] = b3[i];
    __syncthreads();

    const int lane = tid & 31;
    const int wave = tid >> 5;
    const int gid  = blockIdx.x * 2 + wave;      // grid sized exactly B*S/2
    const int b    = gid / NPOINT;
    const int s    = gid % NPOINT;

    _Float16* g  = gbuf[wave];   // [32 x 96]  (stride 96)
    _Float16* h1 = hbuf[wave];   // [32 x 64]  (stride 64)
    _Float16* h2 = gbuf[wave];   // reuse g as [32 x 64] (stride 64)

    // --- gather: lane j builds row j of g = [dxyz(3) | feats(64) | pad(29)] ---
    const float cx = xyz[((size_t)b * NPTS + s) * 3 + 0];
    const float cy = xyz[((size_t)b * NPTS + s) * 3 + 1];
    const float cz = xyz[((size_t)b * NPTS + s) * 3 + 2];
    const int nj = idx_ws[(size_t)gid * NSAMP + lane];
    const float* pj = xyz + ((size_t)b * NPTS + nj) * 3;
    g[lane * 96 + 0] = (_Float16)(pj[0] - cx);
    g[lane * 96 + 1] = (_Float16)(pj[1] - cy);
    g[lane * 96 + 2] = (_Float16)(pj[2] - cz);
    const float* fbase = features + (size_t)b * NCH * NPTS + nj;
#pragma unroll 4
    for (int c = 0; c < NCH; ++c)
        g[lane * 96 + 3 + c] = (_Float16)fbase[(size_t)c * NPTS];
#pragma unroll
    for (int c = 67; c < 96; ++c) g[lane * 96 + c] = (_Float16)0.0f;
    __syncthreads();

    const int half = lane >> 4;

    // --- layer 1: [32x96] x [96x64] -> h1 [32x64], ReLU ---
#pragma unroll
    for (int mt = 0; mt < 2; ++mt) {
#pragma unroll
        for (int nt = 0; nt < 4; ++nt) {
            const float bv = bias1[nt * 16 + (lane & 15)];
            v8f acc;
#pragma unroll
            for (int r = 0; r < 8; ++r) acc[r] = bv;
#pragma unroll
            for (int kt = 0; kt < 3; ++kt) {
                v16h a = load_a_frag(g, 96, mt, kt, lane);
                v16h w = load_b_frag(Wt1, 96, nt, kt, lane);
                acc = __builtin_amdgcn_wmma_f32_16x16x32_f16(
                          false, a, false, w, (short)0, acc, false, false);
            }
#pragma unroll
            for (int r = 0; r < 8; ++r) {
                const float v = fmaxf(acc[r], 0.0f);
                h1[(mt * 16 + half * 8 + r) * 64 + nt * 16 + (lane & 15)] = (_Float16)v;
            }
        }
    }
    __syncthreads();

    // --- layer 2: [32x64] x [64x64] -> h2 [32x64], ReLU ---
#pragma unroll
    for (int mt = 0; mt < 2; ++mt) {
#pragma unroll
        for (int nt = 0; nt < 4; ++nt) {
            const float bv = bias2[nt * 16 + (lane & 15)];
            v8f acc;
#pragma unroll
            for (int r = 0; r < 8; ++r) acc[r] = bv;
#pragma unroll
            for (int kt = 0; kt < 2; ++kt) {
                v16h a = load_a_frag(h1, 64, mt, kt, lane);
                v16h w = load_b_frag(Wt2, 64, nt, kt, lane);
                acc = __builtin_amdgcn_wmma_f32_16x16x32_f16(
                          false, a, false, w, (short)0, acc, false, false);
            }
#pragma unroll
            for (int r = 0; r < 8; ++r) {
                const float v = fmaxf(acc[r], 0.0f);
                h2[(mt * 16 + half * 8 + r) * 64 + nt * 16 + (lane & 15)] = (_Float16)v;
            }
        }
    }
    __syncthreads();

    // --- layer 3: [32x64] x [64x128] + ReLU, then max over the 32 rows ---
#pragma unroll
    for (int nt = 0; nt < 8; ++nt) {
        const float bv = bias3[nt * 16 + (lane & 15)];
        v8f acc0, acc1;
#pragma unroll
        for (int r = 0; r < 8; ++r) { acc0[r] = bv; acc1[r] = bv; }
#pragma unroll
        for (int kt = 0; kt < 2; ++kt) {
            v16h w  = load_b_frag(Wt3, 64, nt, kt, lane);
            v16h a0 = load_a_frag(h2, 64, 0, kt, lane);
            v16h a1 = load_a_frag(h2, 64, 1, kt, lane);
            acc0 = __builtin_amdgcn_wmma_f32_16x16x32_f16(
                       false, a0, false, w, (short)0, acc0, false, false);
            acc1 = __builtin_amdgcn_wmma_f32_16x16x32_f16(
                       false, a1, false, w, (short)0, acc1, false, false);
        }
        // ReLU + column max: init 0 folds in ReLU (all outputs clamped >= 0)
        float v = 0.0f;
#pragma unroll
        for (int r = 0; r < 8; ++r) {
            v = fmaxf(v, acc0[r]);
            v = fmaxf(v, acc1[r]);
        }
        v = fmaxf(v, __shfl_xor(v, 16, 32));   // combine row-halves
        if (lane < 16)
            out_feat[((size_t)b * 128 + nt * 16 + lane) * NPOINT + s] = v;
    }
}

// ---------------------------------------------------------------------------
extern "C" void kernel_launch(void* const* d_in, const int* in_sizes, int n_in,
                              void* d_out, int out_size, void* d_ws, size_t ws_size,
                              hipStream_t stream) {
    const float* xyz      = (const float*)d_in[0];
    const float* features = (const float*)d_in[1];
    const float* W1       = (const float*)d_in[2];
    const float* b1       = (const float*)d_in[3];
    const float* W2       = (const float*)d_in[4];
    const float* b2       = (const float*)d_in[5];
    const float* W3       = (const float*)d_in[6];
    const float* b3       = (const float*)d_in[7];
    // d_in[8] = npoint (4096, compiled in)

    float* out_newxyz   = (float*)d_out;                               // B*S*3
    float* out_feat     = out_newxyz + (size_t)BATCH * NPOINT * 3;     // B*128*S
    int*   out_sampidx  = (int*)(out_feat + (size_t)BATCH * 128 * NPOINT); // B*S
    int*   idx_ws       = (int*)d_ws;                                  // B*S*32 ints

    // Kernel 1: 4 waves/block, one centroid per wave.
    const int total = BATCH * NPOINT;                 // 8192
    ball_query_kernel<<<total / 4, 128, 0, stream>>>(
        xyz, out_newxyz, out_sampidx, idx_ws);

    // Kernel 2: 2 waves/block, one sample per wave.
    group_mlp_kernel<<<total / 2, 64, 0, stream>>>(
        xyz, features, W1, b1, W2, b2, W3, b3, idx_ws, out_feat);
}
